// GlobalAttentionGeneral_5652176962286
// MI455X (gfx1250) — compile-verified, matrix operand order checked by reading, not run on hardware
//
#include <hip/hip_runtime.h>
#include <hip/hip_bf16.h>

typedef __attribute__((ext_vector_type(16))) __bf16 bf16x16;
typedef __attribute__((ext_vector_type(8)))  __bf16 bf16x8;
typedef __attribute__((ext_vector_type(8)))  float  f32x8;

#define NB 32
#define NT 64
#define ND 512
#define NSR 4096
#define GAMMA1 4.0f

static __device__ __forceinline__ f32x8 wmma_bf16(bf16x16 a, bf16x16 b, f32x8 c) {
    return __builtin_amdgcn_wmma_f32_16x16x32_bf16(false, a, false, b, (short)0, c, false, false);
}

static __device__ __forceinline__ bf16x8 cvt8(float4 a, float4 b) {
    bf16x8 h;
    h[0] = (__bf16)a.x; h[1] = (__bf16)a.y; h[2] = (__bf16)a.z; h[3] = (__bf16)a.w;
    h[4] = (__bf16)b.x; h[5] = (__bf16)b.y; h[6] = (__bf16)b.z; h[7] = (__bf16)b.w;
    return h;
}

static __device__ __forceinline__ bf16x16 cat8(bf16x8 lo, bf16x8 hi) {
    return __builtin_shufflevector(lo, hi, 0,1,2,3,4,5,6,7,8,9,10,11,12,13,14,15);
}

// -------------------------------------------------------------------------
// Kernel 1: S = ctx @ q^T  (per batch, 128-row tile), row softmax over T,
// E = exp(GAMMA1 * softmax_T(S)), per-tile column partial sums.
// grid = B*32 blocks, 256 threads (8 waves). Wave w owns rows [16w,16w+16).
// -------------------------------------------------------------------------
__global__ __launch_bounds__(256)
void ga_k1_gemm1_softmax(const float* __restrict__ q,
                         const float* __restrict__ ctx,
                         float* __restrict__ Ew,
                         float* __restrict__ partial) {
    __shared__ __bf16 ldsQ[NT * 32];      // q chunk [t][k(32)] bf16, 4KB
    __shared__ float  ldsColW[8][NT];     // per-wave column partials

    const int b    = blockIdx.x >> 5;
    const int tile = blockIdx.x & 31;
    const int tid  = threadIdx.x;
    const int wave = tid >> 5;
    const int lane = tid & 31;
    const int half = lane >> 4;
    const int ln   = lane & 15;
    const int k0   = half * 8;            // per-lane K phase (ISA A/B layout)

    const int srow = tile * 128 + wave * 16 + ln;              // A-row (M = ln)
    const float* arow = ctx + ((size_t)b * NSR + srow) * ND;
    const float* qb   = q + (size_t)b * NT * ND;

    f32x8 c0 = {}, c1 = {}, c2 = {}, c3 = {};

    for (int kb = 0; kb < ND; kb += 32) {
        __syncthreads();
        // stage q[b, :, kb..kb+32) as bf16 into LDS (row-major [t][32])
        {
            const int t  = tid >> 2;
            const int kk = (tid & 3) * 8;
            const float* qp = qb + (size_t)t * ND + kb + kk;
            float4 y0 = ((const float4*)qp)[0];
            float4 y1 = ((const float4*)qp)[1];
            *(bf16x8*)&ldsQ[t * 32 + kk] = cvt8(y0, y1);
        }
        __syncthreads();

        // A fragment straight from global (rows are K-contiguous)
        float4 x0 = ((const float4*)(arow + kb + k0))[0];
        float4 x1 = ((const float4*)(arow + kb + k0))[1];
        float4 x2 = ((const float4*)(arow + kb + 16 + k0))[0];
        float4 x3 = ((const float4*)(arow + kb + 16 + k0))[1];
        bf16x16 a = cat8(cvt8(x0, x1), cvt8(x2, x3));

        // 4 column tiles of q^T from LDS
        {
            const __bf16* bp = &ldsQ[(0 * 16 + ln) * 32 + k0];
            bf16x16 bb = cat8(*(const bf16x8*)bp, *(const bf16x8*)(bp + 16));
            c0 = wmma_bf16(a, bb, c0);
        }
        {
            const __bf16* bp = &ldsQ[(1 * 16 + ln) * 32 + k0];
            bf16x16 bb = cat8(*(const bf16x8*)bp, *(const bf16x8*)(bp + 16));
            c1 = wmma_bf16(a, bb, c1);
        }
        {
            const __bf16* bp = &ldsQ[(2 * 16 + ln) * 32 + k0];
            bf16x16 bb = cat8(*(const bf16x8*)bp, *(const bf16x8*)(bp + 16));
            c2 = wmma_bf16(a, bb, c2);
        }
        {
            const __bf16* bp = &ldsQ[(3 * 16 + ln) * 32 + k0];
            bf16x16 bb = cat8(*(const bf16x8*)bp, *(const bf16x8*)(bp + 16));
            c3 = wmma_bf16(a, bb, c3);
        }
    }

    // Row softmax over T=64 (row m = 8*half + r lives in one 16-lane half),
    // then E = exp(GAMMA1 * p). Values of GAMMA1*p are in (0,4] -> the SR
    // softmax later needs no max subtraction.
    float ex0[8], ex1[8], ex2[8], ex3[8];
#pragma unroll
    for (int r = 0; r < 8; ++r) {
        float m = fmaxf(fmaxf(c0[r], c1[r]), fmaxf(c2[r], c3[r]));
        m = fmaxf(m, __shfl_xor(m, 1, 32));
        m = fmaxf(m, __shfl_xor(m, 2, 32));
        m = fmaxf(m, __shfl_xor(m, 4, 32));
        m = fmaxf(m, __shfl_xor(m, 8, 32));
        float e0 = __expf(c0[r] - m);
        float e1 = __expf(c1[r] - m);
        float e2 = __expf(c2[r] - m);
        float e3 = __expf(c3[r] - m);
        float s = e0 + e1 + e2 + e3;
        s += __shfl_xor(s, 1, 32);
        s += __shfl_xor(s, 2, 32);
        s += __shfl_xor(s, 4, 32);
        s += __shfl_xor(s, 8, 32);
        float g = GAMMA1 / s;
        ex0[r] = __expf(e0 * g);
        ex1[r] = __expf(e1 * g);
        ex2[r] = __expf(e2 * g);
        ex3[r] = __expf(e3 * g);
    }

    // write E and accumulate deterministic column partials
    const size_t ebase = ((size_t)b * NSR + (size_t)tile * 128 + wave * 16 + 8 * half) * NT + ln;
    float cp0 = 0.f, cp1 = 0.f, cp2 = 0.f, cp3 = 0.f;
#pragma unroll
    for (int r = 0; r < 8; ++r) {
        Ew[ebase + (size_t)r * NT +  0] = ex0[r]; cp0 += ex0[r];
        Ew[ebase + (size_t)r * NT + 16] = ex1[r]; cp1 += ex1[r];
        Ew[ebase + (size_t)r * NT + 32] = ex2[r]; cp2 += ex2[r];
        Ew[ebase + (size_t)r * NT + 48] = ex3[r]; cp3 += ex3[r];
    }
    cp0 += __shfl_xor(cp0, 16, 32);
    cp1 += __shfl_xor(cp1, 16, 32);
    cp2 += __shfl_xor(cp2, 16, 32);
    cp3 += __shfl_xor(cp3, 16, 32);
    if (half == 0) {
        ldsColW[wave][ 0 + ln] = cp0;
        ldsColW[wave][16 + ln] = cp1;
        ldsColW[wave][32 + ln] = cp2;
        ldsColW[wave][48 + ln] = cp3;
    }
    __syncthreads();
    if (tid < NT) {
        float s = 0.f;
#pragma unroll
        for (int w = 0; w < 8; ++w) s += ldsColW[w][tid];
        partial[((size_t)b * 32 + tile) * NT + tid] = s;
    }
}

// -------------------------------------------------------------------------
// Kernel 2: colsum over 32 tile partials, normalize E -> attn (output 2).
// grid = B*32, 256 threads. Deterministic (fixed-order sums).
// -------------------------------------------------------------------------
__global__ __launch_bounds__(256)
void ga_k2_colnorm(const float* __restrict__ Ew,
                   const float* __restrict__ partial,
                   float* __restrict__ attn_out) {
    __shared__ float cs[NT];
    const int b    = blockIdx.x >> 5;
    const int tile = blockIdx.x & 31;
    const int tid  = threadIdx.x;
    if (tid < NT) {
        float s = 0.f;
#pragma unroll
        for (int tl = 0; tl < 32; ++tl)
            s += partial[((size_t)b * 32 + tl) * NT + tid];
        cs[tid] = s;
    }
    __syncthreads();
    const size_t base = (size_t)b * NSR * NT + (size_t)tile * 128 * NT;
    for (int i = tid; i < 128 * NT; i += 256)
        attn_out[base + i] = Ew[base + i] / cs[i & 63];
}

// -------------------------------------------------------------------------
// Kernel 3: weighted[b,d,t] = sum_s ctx[b,s,d] * attn[b,s,t].
// grid = B*8 blocks (64-wide d-chunks), 256 threads. Both operands are
// strided along K=s, so stage 32-s chunks in LDS transposed (pitch 40
// halves = 80B: 16B-aligned b128 fragment loads, conflict-free).
// -------------------------------------------------------------------------
__global__ __launch_bounds__(256)
void ga_k3_gemm2(const float* __restrict__ ctx,
                 const float* __restrict__ attn,
                 float* __restrict__ wout) {
    __shared__ __bf16 ldsA[64 * 40];   // ctx^T chunk: [d_local][s_local], pitch 40
    __shared__ __bf16 ldsB[64 * 40];   // attn^T chunk: [t][s_local], pitch 40

    const int b   = blockIdx.x >> 3;
    const int d0  = (blockIdx.x & 7) * 64;
    const int tid = threadIdx.x;
    const int wave = tid >> 5;
    const int lane = tid & 31;
    const int half = lane >> 4;
    const int ln   = lane & 15;
    const int k0   = half * 8;
    const int rg   = wave & 3;         // d row-group (16 rows)
    const int cp   = wave >> 2;        // t column pair (2 x 16 cols)

    const float* attn_b = attn + (size_t)b * NSR * NT;

    f32x8 acc0 = {}, acc1 = {};

    const int ss = tid >> 3;           // staging: s_local 0..31
    const int c8 = (tid & 7) << 3;     // staging: 8-wide column group

    for (int s0 = 0; s0 < NSR; s0 += 32) {
        __syncthreads();
        // stage ctx[b, s0+ss, d0+c8 .. +8) transposed as bf16
        {
            const float* cr = ctx + ((size_t)b * NSR + s0 + ss) * ND + d0 + c8;
            float4 u0 = ((const float4*)cr)[0];
            float4 u1 = ((const float4*)cr)[1];
            ldsA[(c8 + 0) * 40 + ss] = (__bf16)u0.x;
            ldsA[(c8 + 1) * 40 + ss] = (__bf16)u0.y;
            ldsA[(c8 + 2) * 40 + ss] = (__bf16)u0.z;
            ldsA[(c8 + 3) * 40 + ss] = (__bf16)u0.w;
            ldsA[(c8 + 4) * 40 + ss] = (__bf16)u1.x;
            ldsA[(c8 + 5) * 40 + ss] = (__bf16)u1.y;
            ldsA[(c8 + 6) * 40 + ss] = (__bf16)u1.z;
            ldsA[(c8 + 7) * 40 + ss] = (__bf16)u1.w;
            const float* pr = attn_b + (size_t)(s0 + ss) * NT + c8;
            float4 w0 = ((const float4*)pr)[0];
            float4 w1 = ((const float4*)pr)[1];
            ldsB[(c8 + 0) * 40 + ss] = (__bf16)w0.x;
            ldsB[(c8 + 1) * 40 + ss] = (__bf16)w0.y;
            ldsB[(c8 + 2) * 40 + ss] = (__bf16)w0.z;
            ldsB[(c8 + 3) * 40 + ss] = (__bf16)w0.w;
            ldsB[(c8 + 4) * 40 + ss] = (__bf16)w1.x;
            ldsB[(c8 + 5) * 40 + ss] = (__bf16)w1.y;
            ldsB[(c8 + 6) * 40 + ss] = (__bf16)w1.z;
            ldsB[(c8 + 7) * 40 + ss] = (__bf16)w1.w;
        }
        __syncthreads();

        const __bf16* ap = &ldsA[(16 * rg + ln) * 40 + k0];
        bf16x16 a = cat8(*(const bf16x8*)ap, *(const bf16x8*)(ap + 16));
        {
            const __bf16* bp = &ldsB[(32 * cp + 0 + ln) * 40 + k0];
            bf16x16 bb = cat8(*(const bf16x8*)bp, *(const bf16x8*)(bp + 16));
            acc0 = wmma_bf16(a, bb, acc0);
        }
        {
            const __bf16* bp = &ldsB[(32 * cp + 16 + ln) * 40 + k0];
            bf16x16 bb = cat8(*(const bf16x8*)bp, *(const bf16x8*)(bp + 16));
            acc1 = wmma_bf16(a, bb, acc1);
        }
    }

#pragma unroll
    for (int r = 0; r < 8; ++r) {
        const int d = d0 + 16 * rg + 8 * half + r;
        const int t0 = 32 * cp + ln;
        wout[((size_t)b * ND + d) * NT + t0 + 0]  = acc0[r];
        wout[((size_t)b * ND + d) * NT + t0 + 16] = acc1[r];
    }
}

extern "C" void kernel_launch(void* const* d_in, const int* in_sizes, int n_in,
                              void* d_out, int out_size, void* d_ws, size_t ws_size,
                              hipStream_t stream) {
    const float* q   = (const float*)d_in[0];   // [B, T, D]
    const float* ctx = (const float*)d_in[1];   // [B, H, W, D] == [B, SR, D]

    float* wout     = (float*)d_out;                              // [B, D, T]
    float* attn_out = (float*)d_out + (size_t)NB * ND * NT;       // [B, SR, T] (== attn_img flat)

    float* Ew      = (float*)d_ws;                                // [B, SR, T] fp32
    float* partial = Ew + (size_t)NB * NSR * NT;                  // [B, 32, T]

    ga_k1_gemm1_softmax<<<NB * 32, 256, 0, stream>>>(q, ctx, Ew, partial);
    ga_k2_colnorm<<<NB * 32, 256, 0, stream>>>(Ew, partial, attn_out);
    ga_k3_gemm2<<<NB * 8, 256, 0, stream>>>(ctx, attn_out, wout);
}
